// IlluminatorAttention_90735479095685
// MI455X (gfx1250) — matrix operand order repaired and checked by reading, not compiled, not run on hardware
//
#include <hip/hip_runtime.h>

#define BDIM 2
#define SEQ  2048
#define DMOD 2048
#define NH   16
#define HD   128
#define D3   (3 * DMOD)
#define QSCALE 0.08838834764831845f /* 1/sqrt(128) */

typedef __attribute__((ext_vector_type(16))) __bf16 v16bf;
typedef __attribute__((ext_vector_type(8)))  float  v8f;
typedef __attribute__((ext_vector_type(4)))  unsigned int v4u;
typedef __attribute__((ext_vector_type(8)))  int v8i;
typedef __attribute__((ext_vector_type(4)))  int v4i;

#if defined(__gfx1250__) && __has_builtin(__builtin_amdgcn_tensor_load_to_lds)
#define HAVE_TDM 1
#else
#define HAVE_TDM 0
#endif

union Frag {
  unsigned int u[8];
  v16bf h;
};

__device__ __forceinline__ unsigned short f2bf(float f) {
  unsigned u = __builtin_bit_cast(unsigned, f);
  return (unsigned short)((u + 0x7FFFu + ((u >> 16) & 1u)) >> 16);
}

__device__ __forceinline__ v8f wmma_bf16(v16bf a, v16bf b, v8f c) {
  return __builtin_amdgcn_wmma_f32_16x16x32_bf16(false, a, false, b, (short)0, c,
                                                 false, false);
}

// A fragment: 16x32 bf16 tile, src row-major [16][ld] (ld even, base dword-aligned).
__device__ __forceinline__ v16bf load_frag_a(const unsigned short* src, int ld) {
  const int lane = threadIdx.x & 31;
  const int row = lane & 15;
  const int kh = lane >> 4;
  Frag f;
#pragma unroll
  for (int j = 0; j < 8; ++j) {
    int k = (j < 4) ? (8 * kh + 2 * j) : (16 + 8 * kh + 2 * (j - 4));
    f.u[j] = *(const unsigned int*)(src + row * ld + k);
  }
  return f.h;
}

// B fragment: 32x16 bf16 tile where element (k,n) is at src[n*ld + k] (n-major).
__device__ __forceinline__ v16bf load_frag_b(const unsigned short* src, int ld) {
  const int lane = threadIdx.x & 31;
  const int col = lane & 15;
  const int kh = lane >> 4;
  Frag f;
#pragma unroll
  for (int j = 0; j < 8; ++j) {
    int k = (j < 4) ? (8 * kh + 2 * j) : (16 + 8 * kh + 2 * (j - 4));
    f.u[j] = *(const unsigned int*)(src + col * ld + k);
  }
  return f.h;
}

#if HAVE_TDM
// TDM: DMA a 2D bf16 tile (tile_d0 contiguous elems x tile_d1 rows, row stride
// stride_el elements) from global into LDS at lds_off (contiguous row-major).
__device__ __forceinline__ void tdm_load_2d(unsigned lds_off, const void* gptr,
                                            unsigned tile_d0, unsigned tile_d1,
                                            unsigned long long stride_el) {
  unsigned long long ga = (unsigned long long)gptr;
  v4u g0;
  g0[0] = 1u;                                           // count=1 (valid user D#)
  g0[1] = lds_off;                                      // LDS byte address
  g0[2] = (unsigned)(ga & 0xFFFFFFFFu);                 // global addr lo
  g0[3] = (unsigned)((ga >> 32) & 0x01FFFFFFu) | (2u << 30);  // addr hi | type=2
  v8i g1;
  g1[0] = (int)(1u << 16);                              // data_size=1 (2 bytes)
  g1[1] = (int)((tile_d0 & 0xFFFFu) << 16);             // tensor_dim0 lo16
  g1[2] = (int)(((tile_d0 >> 16) & 0xFFFFu) |
                ((tile_d1 & 0xFFFFu) << 16));           // dim0 hi | dim1 lo
  g1[3] = (int)(((tile_d1 >> 16) & 0xFFFFu) |
                ((tile_d0 & 0xFFFFu) << 16));           // dim1 hi | tile_dim0
  g1[4] = (int)(tile_d1 & 0xFFFFu);                     // tile_dim1 (tile_dim2=0)
  g1[5] = (int)(stride_el & 0xFFFFFFFFull);             // dim0_stride lo
  g1[6] = (int)((stride_el >> 32) & 0xFFFFull);         // dim0_stride hi
  g1[7] = 0;
  v4i z4 = {0, 0, 0, 0};
#if __clang_major__ >= 23
  v8i z8 = {0, 0, 0, 0, 0, 0, 0, 0};
  __builtin_amdgcn_tensor_load_to_lds(g0, g1, z4, z4, z8, 0);
#else
  __builtin_amdgcn_tensor_load_to_lds(g0, g1, z4, z4, 0);
#endif
}
#endif

__global__ __launch_bounds__(256) void cvt_bf16(const float* __restrict__ src,
                                                unsigned short* __restrict__ dst,
                                                size_t n) {
  size_t i = (size_t)blockIdx.x * 256 + threadIdx.x;
  const size_t stride = (size_t)gridDim.x * 256;
  for (; i < n; i += stride) dst[i] = f2bf(src[i]);
}

// C[M,N] = A[M,K] @ B[K,N] + bias. 128x64 block tile, 4 waves x (32x64),
// A staged by TDM (double-buffered), B staged n-major for dword B-frag loads.
template <bool OUT_BF16>
__global__ __launch_bounds__(128) void gemm_bf16(const unsigned short* __restrict__ A,
                                                 const unsigned short* __restrict__ B,
                                                 const float* __restrict__ bias,
                                                 void* __restrict__ Cout,
                                                 int M, int N, int K) {
  __shared__ unsigned short As[2][128 * 32];  // [buf][m][k] row-major
  __shared__ unsigned short Bt[64 * 32];      // [n][k] n-major
  const int ntiles = N >> 6;
  const int m0 = (blockIdx.x / ntiles) << 7;
  const int n0 = (blockIdx.x % ntiles) << 6;
  const int tid = threadIdx.x;
  const int lane = tid & 31;
  const int w = tid >> 5;

  v8f acc[2][4];
#pragma unroll
  for (int s = 0; s < 2; ++s)
#pragma unroll
    for (int t = 0; t < 4; ++t) acc[s][t] = v8f{};

#if HAVE_TDM
  if (w == 0)
    tdm_load_2d((unsigned)(unsigned long long)&As[0][0], A + (size_t)m0 * K,
                32, 128, (unsigned long long)K);
#endif

  const int nk = K >> 5;
  for (int ik = 0; ik < nk; ++ik) {
    const int k0 = ik << 5;
    const int cur = ik & 1;
    __syncthreads();
#if HAVE_TDM
    if (w == 0) {
      if (ik + 1 < nk) {
        tdm_load_2d((unsigned)(unsigned long long)&As[cur ^ 1][0],
                    A + (size_t)m0 * K + (k0 + 32), 32, 128,
                    (unsigned long long)K);
        __builtin_amdgcn_s_wait_tensorcnt(1);  // chunk ik complete
      } else {
        __builtin_amdgcn_s_wait_tensorcnt(0);
      }
    }
#else
    for (int i = tid; i < 128 * 32; i += 128) {
      int r = i >> 5, k = i & 31;
      As[cur][i] = A[(size_t)(m0 + r) * K + (k0 + k)];
    }
#endif
    for (int i = tid; i < 2048; i += 128) {
      int k = i >> 6, n = i & 63;
      Bt[n * 32 + k] = B[(size_t)(k0 + k) * N + (n0 + n)];
    }
    if (k0 + 32 < K && tid < 32)
      __builtin_prefetch(&B[(size_t)(k0 + 32 + tid) * N + n0], 0, 1);
    __syncthreads();

    v16bf af0 = load_frag_a(&As[cur][(w * 32) * 32], 32);
    v16bf af1 = load_frag_a(&As[cur][(w * 32 + 16) * 32], 32);
#pragma unroll
    for (int t = 0; t < 4; ++t) {
      v16bf bfr = load_frag_b(&Bt[(t * 16) * 32], 32);
      acc[0][t] = wmma_bf16(af0, bfr, acc[0][t]);
      acc[1][t] = wmma_bf16(af1, bfr, acc[1][t]);
    }
  }

  const int nloc = lane & 15;
  const int half = lane >> 4;
#pragma unroll
  for (int s = 0; s < 2; ++s) {
#pragma unroll
    for (int t = 0; t < 4; ++t) {
      int col = n0 + t * 16 + nloc;
      float bv = bias[col];
#pragma unroll
      for (int i = 0; i < 8; ++i) {
        int row = m0 + w * 32 + s * 16 + i + 8 * half;
        float v = acc[s][t][i] + bv;
        if constexpr (OUT_BF16)
          ((unsigned short*)Cout)[(size_t)row * N + col] = f2bf(v);
        else
          ((float*)Cout)[(size_t)row * N + col] = v;
      }
    }
  }
}

// Flash-style causal attention. 4 waves/block; each wave owns 16 q rows of one
// (b,h); K chunks DMA'd by TDM (double-buffered), V staged transposed manually.
__global__ __launch_bounds__(128) void attn_fwd(const unsigned short* __restrict__ qkv,
                                               float* __restrict__ out_w,
                                               unsigned short* __restrict__ ctx,
                                               float* __restrict__ mstat,
                                               float* __restrict__ lstat) {
  __shared__ unsigned short kbuf[2][32 * 128];  // [buf][key][hd]
  __shared__ unsigned short vt[128 * 32];       // [hd][key] (n-major for PV)
  __shared__ unsigned short pbuf[4][16 * 32];   // per-wave [q][key]

  const int qtb = blockIdx.x & (SEQ / 64 - 1);  // SEQ/64 == 32
  const int bh = blockIdx.x >> 5;
  const int h = bh & (NH - 1);
  const int b = bh >> 4;
  const int tid = threadIdx.x;
  const int w = tid >> 5;
  const int lane = tid & 31;
  const int nloc = lane & 15;
  const int half = lane >> 4;
  const int q0 = qtb * 64 + w * 16;
  unsigned short* pw = pbuf[w];

  v16bf qf[4];
  {
    const unsigned short* qbase = qkv + (size_t)(b * SEQ + q0) * D3 + h * HD;
#pragma unroll
    for (int c = 0; c < 4; ++c) qf[c] = load_frag_a(qbase + c * 32, D3);
  }

  v8f o[8];
#pragma unroll
  for (int t = 0; t < 8; ++t) o[t] = v8f{};
  float m_run[8], l_run[8];
#pragma unroll
  for (int i = 0; i < 8; ++i) { m_run[i] = -3.0e38f; l_run[i] = 0.0f; }

  const int nchunks = (qtb * 64 + 64) >> 5;  // block-uniform causal bound
#if HAVE_TDM
  if (w == 0)
    tdm_load_2d((unsigned)(unsigned long long)&kbuf[0][0],
                qkv + (size_t)(b * SEQ) * D3 + DMOD + h * HD, 128, 32,
                (unsigned long long)D3);
#endif

  for (int ci = 0; ci < nchunks; ++ci) {
    const int kb = ci << 5;
    const int cur = ci & 1;
    __syncthreads();
#if HAVE_TDM
    if (w == 0) {
      if (ci + 1 < nchunks) {
        tdm_load_2d((unsigned)(unsigned long long)&kbuf[cur ^ 1][0],
                    qkv + (size_t)(b * SEQ + kb + 32) * D3 + DMOD + h * HD,
                    128, 32, (unsigned long long)D3);
        __builtin_amdgcn_s_wait_tensorcnt(1);
      } else {
        __builtin_amdgcn_s_wait_tensorcnt(0);
      }
    }
#else
    {
      const size_t kbase = (size_t)(b * SEQ + kb) * D3 + DMOD + h * HD;
      unsigned int* dst = (unsigned int*)kbuf[cur];
      for (int i = tid; i < 2048; i += 128) {
        int r = i >> 6, c = i & 63;
        dst[i] = *(const unsigned int*)(qkv + kbase + (size_t)r * D3 + 2 * c);
      }
    }
#endif
    {  // stage V chunk transposed [hd][key]
      const size_t vbase = (size_t)(b * SEQ + kb) * D3 + 2 * DMOD + h * HD;
      for (int i = tid; i < 4096; i += 128) {
        int r = i >> 7, hd = i & 127;
        vt[hd * 32 + r] = qkv[vbase + (size_t)r * D3 + hd];
      }
      if (ci + 1 < nchunks && tid < 32)
        __builtin_prefetch(qkv + vbase + (size_t)(32 + tid) * D3, 0, 1);
    }
    __syncthreads();

    // scores = Q @ K^T for 32 keys (two 16-wide tiles), K-depth HD=128
    v8f s0 = v8f{}, s1 = v8f{};
#pragma unroll
    for (int c = 0; c < 4; ++c) {
      v16bf k0f = load_frag_b(&kbuf[cur][c * 32], 128);
      v16bf k1f = load_frag_b(&kbuf[cur][16 * 128 + c * 32], 128);
      s0 = wmma_bf16(qf[c], k0f, s0);
      s1 = wmma_bf16(qf[c], k1f, s1);
    }

    float corr[8];
#pragma unroll
    for (int i = 0; i < 8; ++i) {
      const int q = q0 + i + 8 * half;
      const int kg0 = kb + nloc;
      const int kg1 = kb + 16 + nloc;
      float a = s0[i] * QSCALE;
      if (kg0 > q) a = -1.0e30f;
      float c2 = s1[i] * QSCALE;
      if (kg1 > q) c2 = -1.0e30f;
      const size_t rowoff = ((size_t)bh * SEQ + q) * SEQ;
      if (kg0 <= q) out_w[rowoff + kg0] = a;  // raw masked scores (lower tri)
      if (kg1 <= q) out_w[rowoff + kg1] = c2;
      float t = fmaxf(a, c2);
      t = fmaxf(t, __shfl_xor(t, 1));
      t = fmaxf(t, __shfl_xor(t, 2));
      t = fmaxf(t, __shfl_xor(t, 4));
      t = fmaxf(t, __shfl_xor(t, 8));
      float mnew = fmaxf(m_run[i], t);
      float cf = __expf(m_run[i] - mnew);
      m_run[i] = mnew;
      corr[i] = cf;
      float p0 = __expf(a - mnew);
      float p1 = __expf(c2 - mnew);
      float rs = p0 + p1;
      rs += __shfl_xor(rs, 1);
      rs += __shfl_xor(rs, 2);
      rs += __shfl_xor(rs, 4);
      rs += __shfl_xor(rs, 8);
      l_run[i] = l_run[i] * cf + rs;
      pw[(i + 8 * half) * 32 + nloc] = f2bf(p0);
      pw[(i + 8 * half) * 32 + 16 + nloc] = f2bf(p1);
    }
#pragma unroll
    for (int t = 0; t < 8; ++t)
#pragma unroll
      for (int i = 0; i < 8; ++i) o[t][i] *= corr[i];
    __syncthreads();

    // O += P @ V (K-depth 32 keys, 8 hd tiles)
    v16bf pf = load_frag_a(pw, 32);
#pragma unroll
    for (int t = 0; t < 8; ++t) {
      v16bf vf = load_frag_b(&vt[(t * 16) * 32], 32);
      o[t] = wmma_bf16(pf, vf, o[t]);
    }
  }

#pragma unroll
  for (int t = 0; t < 8; ++t) {
#pragma unroll
    for (int i = 0; i < 8; ++i) {
      int q = q0 + i + 8 * half;
      float v = o[t][i] / l_run[i];
      ctx[(size_t)(b * SEQ + q) * DMOD + h * HD + t * 16 + nloc] = f2bf(v);
    }
  }
  if (nloc == 0) {
#pragma unroll
    for (int i = 0; i < 8; ++i) {
      int q = q0 + i + 8 * half;
      mstat[(size_t)bh * SEQ + q] = m_run[i];
      lstat[(size_t)bh * SEQ + q] = l_run[i];
    }
  }
}

// Normalize stored raw scores in place: w = exp(s - m)/l (0 above diagonal).
__global__ __launch_bounds__(256) void norm_w(const float* __restrict__ mstat,
                                              const float* __restrict__ lstat,
                                              float* __restrict__ w) {
  const size_t total = (size_t)BDIM * NH * SEQ * SEQ;
  size_t idx = (size_t)blockIdx.x * 256 + threadIdx.x;
  if (idx >= total) return;
  size_t k = idx & (SEQ - 1);
  size_t q = (idx >> 11) & (SEQ - 1);
  size_t bh = idx >> 22;
  float val = 0.0f;
  if (k <= q) {
    float sc = w[idx];
    val = __expf(sc - mstat[bh * SEQ + q]) / lstat[bh * SEQ + q];
  }
  w[idx] = val;
}

extern "C" void kernel_launch(void* const* d_in, const int* in_sizes, int n_in,
                              void* d_out, int out_size, void* d_ws, size_t ws_size,
                              hipStream_t stream) {
  (void)in_sizes; (void)n_in; (void)out_size; (void)ws_size;
  const float* x = (const float*)d_in[0];
  const float* w_attn = (const float*)d_in[1];
  const float* b_attn = (const float*)d_in[2];
  const float* w_proj = (const float*)d_in[3];
  const float* b_proj = (const float*)d_in[4];

  float* out_ctx = (float*)d_out;                      // [B,S,D]
  float* out_w = out_ctx + (size_t)BDIM * SEQ * DMOD;  // [B,H,S,S]

  char* ws = (char*)d_ws;
  size_t off = 0;
  auto alloc = [&](size_t bytes) {
    char* p = ws + off;
    off = (off + bytes + 255) & ~(size_t)255;
    return p;
  };
  const size_t nx = (size_t)BDIM * SEQ * DMOD;
  const size_t nwa = (size_t)DMOD * D3;
  const size_t nwp = (size_t)DMOD * DMOD;
  const size_t nqkv = (size_t)BDIM * SEQ * D3;

  unsigned short* x_bf = (unsigned short*)alloc(nx * 2);
  unsigned short* wa_bf = (unsigned short*)alloc(nwa * 2);
  unsigned short* wp_bf = (unsigned short*)alloc(nwp * 2);
  unsigned short* qkv_bf = (unsigned short*)alloc(nqkv * 2);
  unsigned short* ctx_bf = (unsigned short*)alloc(nx * 2);
  float* mstat = (float*)alloc((size_t)BDIM * NH * SEQ * 4);
  float* lstat = (float*)alloc((size_t)BDIM * NH * SEQ * 4);

  cvt_bf16<<<2048, 256, 0, stream>>>(x, x_bf, nx);
  cvt_bf16<<<2048, 256, 0, stream>>>(w_attn, wa_bf, nwa);
  cvt_bf16<<<2048, 256, 0, stream>>>(w_proj, wp_bf, nwp);

  // qkv = x @ w_attn + b_attn : [4096 x 2048] @ [2048 x 6144]
  gemm_bf16<true><<<(BDIM * SEQ / 128) * (D3 / 64), 128, 0, stream>>>(
      x_bf, wa_bf, b_attn, qkv_bf, BDIM * SEQ, D3, DMOD);

  attn_fwd<<<BDIM * NH * (SEQ / 64), 128, 0, stream>>>(qkv_bf, out_w, ctx_bf,
                                                       mstat, lstat);

  const size_t totw = (size_t)BDIM * NH * SEQ * SEQ;
  norm_w<<<(unsigned)((totw + 255) / 256), 256, 0, stream>>>(mstat, lstat, out_w);

  // out = ctx @ w_proj + b_proj : [4096 x 2048] @ [2048 x 2048]
  gemm_bf16<false><<<(BDIM * SEQ / 128) * (DMOD / 64), 128, 0, stream>>>(
      ctx_bf, wp_bf, b_proj, out_ctx, BDIM * SEQ, DMOD, DMOD);
}
